// CausalSelfAttention_32882269618512
// MI455X (gfx1250) — matrix operand order repaired and checked by reading, not compiled, not run on hardware
//
#include <hip/hip_runtime.h>
#include <stdint.h>

// Causal self-attention forward, CDNA5 (gfx1250), bf16 WMMA pipeline.
// cast/transpose -> QKV GEMM (async-LDS double-buffered WMMA) -> flash attention (WMMA)
// -> proj GEMM (async-LDS WMMA).

#define D_MODEL   1024
#define N_HEADS   16
#define HEAD_DIM  64
#define BATCH     4
#define SEQ       2048
#define M_TOT     (BATCH * SEQ)   // 8192 rows

// ---- gfx1250 async-copy feature detection (bridge doc: names differ across toolchains)
#if defined(__has_builtin)
#  if __has_builtin(__builtin_amdgcn_global_load_async_to_lds_b128)
#    define HAVE_ASYNC_LDS 1
#  endif
#  if __has_builtin(__builtin_amdgcn_s_wait_asynccnt)
#    define HAVE_WAIT_ASYNC 1
#  endif
#endif

typedef __attribute__((ext_vector_type(16))) __bf16 v16bf;
typedef __attribute__((ext_vector_type(8)))  float  v8f;
typedef __attribute__((ext_vector_type(4)))  int    v4i;
typedef __attribute__((address_space(1))) v4i* gv4i_p;   // global int4*
typedef __attribute__((address_space(3))) v4i* lv4i_p;   // LDS int4*

union FragU { v16bf v; uint4 u[2]; };

__device__ __forceinline__ unsigned short f2bf(float f) {
  uint32_t u = __float_as_uint(f);
  u += 0x7fffu + ((u >> 16) & 1u);          // round-to-nearest-even
  return (unsigned short)(u >> 16);
}

// Load a 16-element bf16 fragment as two 16-byte chunks (global or LDS; addrspace inferred).
__device__ __forceinline__ v16bf load_frag(const unsigned short* p0, const unsigned short* p1) {
  FragU f;
  f.u[0] = *reinterpret_cast<const uint4*>(p0);
  f.u[1] = *reinterpret_cast<const uint4*>(p1);
  return f.v;
}

__device__ __forceinline__ v8f wmma_bf16(v16bf a, v16bf b, v8f c) {
  return __builtin_amdgcn_wmma_f32_16x16x32_bf16(false, a, false, b, (short)0, c, false, false);
}

// 16B global -> LDS copy: async DMA (ASYNCcnt) if available, else reg round-trip.
__device__ __forceinline__ void cp_g2l_b128(const unsigned short* gp, unsigned short* lp) {
#if HAVE_ASYNC_LDS
  v4i* g = (v4i*)const_cast<unsigned short*>(gp);
  v4i* l = (v4i*)lp;
  __builtin_amdgcn_global_load_async_to_lds_b128((gv4i_p)g, (lv4i_p)l, 0, 0);
#else
  *reinterpret_cast<uint4*>(lp) = *reinterpret_cast<const uint4*>(gp);
#endif
}

__device__ __forceinline__ void wait_async_le6() {
#if HAVE_WAIT_ASYNC
  __builtin_amdgcn_s_wait_asynccnt(6);
#elif HAVE_ASYNC_LDS
  asm volatile("s_wait_asynccnt 6" ::: "memory");
#endif
}
__device__ __forceinline__ void wait_async_le0() {
#if HAVE_WAIT_ASYNC
  __builtin_amdgcn_s_wait_asynccnt(0);
#elif HAVE_ASYNC_LDS
  asm volatile("s_wait_asynccnt 0" ::: "memory");
#endif
}

// ---------------------------------------------------------------- cast kernels
__global__ void cast_f32_bf16(const float* __restrict__ in, unsigned short* __restrict__ out, int n) {
  int i = (blockIdx.x * blockDim.x + threadIdx.x) * 4;
  if (i + 3 < n) {
    float4 f = *reinterpret_cast<const float4*>(in + i);
    out[i + 0] = f2bf(f.x); out[i + 1] = f2bf(f.y);
    out[i + 2] = f2bf(f.z); out[i + 3] = f2bf(f.w);
  } else {
    for (int j = i; j < n; ++j) out[j] = f2bf(in[j]);
  }
}

// in: [K][N] f32 row-major  ->  out: [N][K] bf16 (transposed)
__global__ void tcast_f32_bf16(const float* __restrict__ in, unsigned short* __restrict__ out,
                               int K, int N) {
  long i = (long)blockIdx.x * blockDim.x + threadIdx.x;
  if (i < (long)K * N) {
    int k = (int)(i / N), n = (int)(i % N);
    out[(size_t)n * K + k] = f2bf(in[i]);
  }
}

// ---------------------------------------------------------------- GEMM
// C[M,N] = A[M,Kd] (bf16 row-major) x BT[N,Kd]^T (bf16 pre-transposed) + bias.
// Block: 256 threads = 8 waves; block tile 256(M) x 128(N); wave tile 64x64; K-step 32.
// A/B tiles double-buffered in LDS via async global->LDS DMA; fragments fed by ds_load_b128.
// LDS row stride padded to 40 elems (80 B = 20 banks): 16 fragment rows land on 16
// distinct bank groups -> conflict-free ds reads.
#define GK       32
#define LDSTR    40                         // 32 + 8 pad (keeps 16B alignment: 80B rows)
#define A_ROWS   256
#define B_ROWS   128

__global__ __launch_bounds__(256)
void gemm_bf16_wmma(const unsigned short* __restrict__ A,
                    const unsigned short* __restrict__ BT,
                    const float* __restrict__ bias,
                    int M, int N, int Kd, int mode,
                    unsigned short* __restrict__ qo,
                    unsigned short* __restrict__ ko,
                    unsigned short* __restrict__ vo,
                    float* __restrict__ outf) {
  __shared__ unsigned short As[2][A_ROWS * LDSTR];   // 2 x 20 KB
  __shared__ unsigned short Bs[2][B_ROWS * LDSTR];   // 2 x 10 KB

  const int tid  = threadIdx.x;
  const int lane = tid & 31;
  const int w    = tid >> 5;
  const int half = lane >> 4;        // 0 | 1
  const int lr   = lane & 15;
  const int waveM = w & 3, waveN = w >> 2;
  const long m0 = (long)blockIdx.y * 256 + waveM * 64;
  const long n0 = (long)blockIdx.x * 128 + waveN * 64;
  const long mBlk = (long)blockIdx.y * 256;
  const long nBlk = (long)blockIdx.x * 128;

  // stage one 32-deep K slice of A (256 rows) and B (128 rows) into LDS buffer `buf`
  auto stage = [&](int buf, int k0) {
#pragma unroll
    for (int q = 0; q < 4; ++q) {            // A: 1024 16B-chunks / 256 threads
      int c   = tid * 4 + q;
      int row = c >> 2, kc = (c & 3) * 8;
      cp_g2l_b128(A + (size_t)(mBlk + row) * Kd + k0 + kc,
                  &As[buf][row * LDSTR + kc]);
    }
#pragma unroll
    for (int q = 0; q < 2; ++q) {            // B: 512 16B-chunks / 256 threads
      int c   = tid * 2 + q;
      int row = c >> 2, kc = (c & 3) * 8;
      cp_g2l_b128(BT + (size_t)(nBlk + row) * Kd + k0 + kc,
                  &Bs[buf][row * LDSTR + kc]);
    }
  };

  v8f acc[4][4];
#pragma unroll
  for (int i = 0; i < 4; ++i)
#pragma unroll
    for (int j = 0; j < 4; ++j)
      acc[i][j] = (v8f){0.f, 0.f, 0.f, 0.f, 0.f, 0.f, 0.f, 0.f};

  stage(0, 0);                               // prologue: 6 async ops outstanding

  for (int k0 = 0; k0 < Kd; k0 += GK) {
    const int buf = (k0 >> 5) & 1;
    if (k0 + GK < Kd) {                      // overlap next tile's DMA with this tile's math
      stage(buf ^ 1, k0 + GK);
      wait_async_le6();                      // in-order completion => current tile landed
    } else {
      wait_async_le0();
    }
    __syncthreads();                         // everyone's stage visible

    v16bf af[4], bf[4];
#pragma unroll
    for (int i = 0; i < 4; ++i) {
      // A fragment: lane row = tile_m + lr; K chunks [8h,+8) and [16+8h,+8)
      const unsigned short* p = &As[buf][(waveM * 64 + 16 * i + lr) * LDSTR];
      af[i] = load_frag(p + 8 * half, p + 16 + 8 * half);
    }
#pragma unroll
    for (int j = 0; j < 4; ++j) {
      // B fragment: lane col n = lr -> row of BT; K chunks [8h,+8) and [16+8h,+8)
      const unsigned short* p = &Bs[buf][(waveN * 64 + 16 * j + lr) * LDSTR];
      bf[j] = load_frag(p + 8 * half, p + 16 + 8 * half);
    }
#pragma unroll
    for (int i = 0; i < 4; ++i)
#pragma unroll
      for (int j = 0; j < 4; ++j)
        acc[i][j] = wmma_bf16(af[i], bf[j], acc[i][j]);

    __syncthreads();                         // all reads done before buf^1 is overwritten
  }

#pragma unroll
  for (int i = 0; i < 4; ++i) {
#pragma unroll
    for (int j = 0; j < 4; ++j) {
#pragma unroll
      for (int v = 0; v < 8; ++v) {
        const int m = (int)m0 + 16 * i + v + 8 * half;  // C layout: row = v + 8*(lane/16)
        const int n = (int)n0 + 16 * j + lr;            //           col = lane%16
        float val = acc[i][j][v] + bias[n];
        if (mode == 0) {
          const int b = m >> 11, s = m & (SEQ - 1);
          const int sec = n >> 10, n1 = n & 1023;
          const int h = n1 >> 6, d = n1 & 63;
          const size_t bh = (size_t)(b * N_HEADS + h);
          if (sec == 0)       qo[(bh * SEQ + s) * HEAD_DIM + d] = f2bf(val * 0.125f); // 1/sqrt(64)
          else if (sec == 1)  ko[(bh * SEQ + s) * HEAD_DIM + d] = f2bf(val);
          else                vo[(bh * HEAD_DIM + d) * SEQ + s] = f2bf(val);          // V transposed
        } else {
          outf[(size_t)m * N + n] = val;
        }
      }
    }
  }
}

// ---------------------------------------------------------------- flash attention
// One block = 4 waves = 64 query rows of one (b,h). Each wave owns 16 q-rows,
// streams keys in 32-wide tiles with online softmax. Q pre-scaled, V^T layout.
__global__ __launch_bounds__(128)
void flash_attn(const unsigned short* __restrict__ Q,
                const unsigned short* __restrict__ K,
                const unsigned short* __restrict__ VT,
                unsigned short* __restrict__ ctx) {
  __shared__ unsigned short pl[4][16 * 32];   // per-wave P staging (C-layout -> A-layout)

  const int lane = threadIdx.x & 31;
  const int w    = threadIdx.x >> 5;
  const int half = lane >> 4;
  const int lr   = lane & 15;
  const int bh = blockIdx.y;                  // b*16 + h
  const int b  = bh >> 4, h = bh & 15;
  const int q_lo = blockIdx.x * 64 + w * 16;

  const unsigned short* Qb = Q  + (size_t)bh * SEQ * HEAD_DIM;
  const unsigned short* Kb = K  + (size_t)bh * SEQ * HEAD_DIM;
  const unsigned short* Vb = VT + (size_t)bh * HEAD_DIM * SEQ;

  // Q fragments (A 16x32) over head_dim split into two 32-chunks
  v16bf qf[2];
#pragma unroll
  for (int c = 0; c < 2; ++c) {
    const unsigned short* p = Qb + (size_t)(q_lo + lr) * HEAD_DIM + c * 32;
    qf[c] = load_frag(p + 8 * half, p + 16 + 8 * half);
  }

  v8f o[4];
#pragma unroll
  for (int t = 0; t < 4; ++t) o[t] = (v8f){0.f, 0.f, 0.f, 0.f, 0.f, 0.f, 0.f, 0.f};
  float mst[8], lst[8];
#pragma unroll
  for (int v = 0; v < 8; ++v) { mst[v] = -3.0e38f; lst[v] = 0.f; }

  for (int kv = 0; kv < q_lo + 16; kv += 32) {
    if (kv + 32 < q_lo + 16) {                // hint next K/V tiles toward the caches
      __builtin_prefetch(Kb + (size_t)(kv + 32 + lr) * HEAD_DIM, 0, 1);
      __builtin_prefetch(Vb + (size_t)lr * SEQ + kv + 32, 0, 1);
    }

    // ---- S = (Q*scale) @ K^T for two 16-wide key tiles
    v8f s[2];
#pragma unroll
    for (int t = 0; t < 2; ++t) {
      const unsigned short* kp = Kb + (size_t)(kv + 16 * t + lr) * HEAD_DIM;
      v16bf kf0 = load_frag(kp + 8 * half,      kp + 16 + 8 * half);        // hd 0..31
      v16bf kf1 = load_frag(kp + 32 + 8 * half, kp + 48 + 8 * half);       // hd 32..63
      v8f z = (v8f){0.f, 0.f, 0.f, 0.f, 0.f, 0.f, 0.f, 0.f};
      z    = wmma_bf16(qf[0], kf0, z);
      s[t] = wmma_bf16(qf[1], kf1, z);
    }

    // ---- causal mask + online softmax (row m = v + 8*half, col n = kv + 16t + lr)
#pragma unroll
    for (int v = 0; v < 8; ++v) {
      const int qi = q_lo + v + 8 * half;
#pragma unroll
      for (int t = 0; t < 2; ++t) {
        const int n = kv + 16 * t + lr;
        if (n > qi) s[t][v] = -3.0e38f;
      }
      float mx = fmaxf(s[0][v], s[1][v]);
#pragma unroll
      for (int msk = 1; msk < 16; msk <<= 1) mx = fmaxf(mx, __shfl_xor(mx, msk, 32));
      const float mnew  = fmaxf(mst[v], mx);
      const float alpha = __expf(mst[v] - mnew);
      mst[v] = mnew;
      const float p0 = __expf(s[0][v] - mnew);
      const float p1 = __expf(s[1][v] - mnew);
      s[0][v] = p0; s[1][v] = p1;
      float rs = p0 + p1;
#pragma unroll
      for (int msk = 1; msk < 16; msk <<= 1) rs += __shfl_xor(rs, msk, 32);
      lst[v] = lst[v] * alpha + rs;
#pragma unroll
      for (int dt = 0; dt < 4; ++dt) o[dt][v] *= alpha;
    }

    // ---- P: C-layout regs -> LDS [16 q][32 kv] (bf16), reload as A-fragment
    unsigned short* myp = pl[w];
#pragma unroll
    for (int v = 0; v < 8; ++v)
#pragma unroll
      for (int t = 0; t < 2; ++t)
        myp[(v + 8 * half) * 32 + 16 * t + lr] = f2bf(s[t][v]);

    const unsigned short* pr = myp + lr * 32;
    v16bf pf = load_frag(pr + 8 * half, pr + 16 + 8 * half);

    // ---- O += P @ V  (V^T rows are contiguous key-runs)
#pragma unroll
    for (int dt = 0; dt < 4; ++dt) {
      const unsigned short* vp = Vb + (size_t)(16 * dt + lr) * SEQ + kv;
      v16bf vf = load_frag(vp + 8 * half, vp + 16 + 8 * half);
      o[dt] = wmma_bf16(pf, vf, o[dt]);
    }
  }

  // ---- normalize and write context [B,S,D] bf16 (A-matrix layout for proj GEMM)
#pragma unroll
  for (int v = 0; v < 8; ++v) {
    const float inv = 1.f / lst[v];
    const int   s_idx = q_lo + v + 8 * half;
#pragma unroll
    for (int dt = 0; dt < 4; ++dt)
      ctx[(size_t)(b * SEQ + s_idx) * D_MODEL + h * HEAD_DIM + 16 * dt + lr] =
          f2bf(o[dt][v] * inv);
  }
}

// ---------------------------------------------------------------- launch
extern "C" void kernel_launch(void* const* d_in, const int* in_sizes, int n_in,
                              void* d_out, int out_size, void* d_ws, size_t ws_size,
                              hipStream_t stream) {
  const float* x      = (const float*)d_in[0];
  const float* qkv_w  = (const float*)d_in[1];
  const float* qkv_b  = (const float*)d_in[2];
  const float* proj_w = (const float*)d_in[3];
  const float* proj_b = (const float*)d_in[4];
  float* out = (float*)d_out;

  char* ws = (char*)d_ws;
  size_t off = 0;
  auto take = [&](size_t bytes) -> char* {
    char* p = ws + off;
    off += (bytes + 255) & ~(size_t)255;
    return p;
  };
  unsigned short* xb  = (unsigned short*)take((size_t)M_TOT * D_MODEL * 2);       // x bf16
  unsigned short* wqt = (unsigned short*)take((size_t)3 * D_MODEL * D_MODEL * 2); // qkv_w^T
  unsigned short* wpt = (unsigned short*)take((size_t)D_MODEL * D_MODEL * 2);     // proj_w^T
  unsigned short* Qw  = (unsigned short*)take((size_t)M_TOT * D_MODEL * 2);       // [B,H,S,64]
  unsigned short* Kw  = (unsigned short*)take((size_t)M_TOT * D_MODEL * 2);       // [B,H,S,64]
  unsigned short* VTw = (unsigned short*)take((size_t)M_TOT * D_MODEL * 2);       // [B,H,64,S]
  unsigned short* ctx = (unsigned short*)take((size_t)M_TOT * D_MODEL * 2);       // [B,S,D]
  (void)ws_size; (void)in_sizes; (void)n_in; (void)out_size;

  { // casts
    int n = M_TOT * D_MODEL;
    cast_f32_bf16<<<(n / 4 + 255) / 256, 256, 0, stream>>>(x, xb, n);
    int nw = 3 * D_MODEL * D_MODEL;
    tcast_f32_bf16<<<(nw + 255) / 256, 256, 0, stream>>>(qkv_w, wqt, D_MODEL, 3 * D_MODEL);
    int np = D_MODEL * D_MODEL;
    tcast_f32_bf16<<<(np + 255) / 256, 256, 0, stream>>>(proj_w, wpt, D_MODEL, D_MODEL);
  }

  // QKV projection: [8192,1024] x [1024,3072]
  gemm_bf16_wmma<<<dim3(3 * D_MODEL / 128, M_TOT / 256), 256, 0, stream>>>(
      xb, wqt, qkv_b, M_TOT, 3 * D_MODEL, D_MODEL, /*mode=*/0, Qw, Kw, VTw, nullptr);

  // Flash attention: 64 q-rows per block, one (b,h) per blockIdx.y
  flash_attn<<<dim3(SEQ / 64, BATCH * N_HEADS), 128, 0, stream>>>(Qw, Kw, VTw, ctx);

  // Output projection: [8192,1024] x [1024,1024] -> f32 out
  gemm_bf16_wmma<<<dim3(D_MODEL / 128, M_TOT / 256), 256, 0, stream>>>(
      ctx, wpt, proj_b, M_TOT, D_MODEL, D_MODEL, /*mode=*/1, nullptr, nullptr, nullptr, out);
}
